// LabelPropagation_15006615733040
// MI455X (gfx1250) — matrix operand order repaired and checked by reading, not compiled, not run on hardware
//
#include <hip/hip_runtime.h>
#include <hip/hip_bf16.h>

typedef __attribute__((ext_vector_type(16))) __bf16 v16bf;
typedef __attribute__((ext_vector_type(8)))  __bf16 v8bf;
typedef __attribute__((ext_vector_type(8)))  float  v8f;
typedef __attribute__((ext_vector_type(4)))  unsigned int v4u;
typedef __attribute__((ext_vector_type(8)))  int    v8i;
typedef __attribute__((ext_vector_type(4)))  int    v4i;

#define N_SUPPORT 4096
#define N_QUERY   4096
#define N_ALL     8192
#define D_IN      512
#define D_EMB     256
#define N_CLASSES 64
#define ITERS     10
#define KT        64      // K-tile depth staged in LDS per pipeline stage

#if defined(__HIP_DEVICE_COMPILE__) && __has_builtin(__builtin_amdgcn_tensor_load_to_lds) && __has_builtin(__builtin_amdgcn_s_wait_tensorcnt)
#define HAVE_TDM 1
#endif

#ifdef HAVE_TDM
// Build a 2D TDM descriptor (D#) and issue tensor_load_to_lds.
// All inputs wave-uniform. data_size = 2 bytes (bf16).
__device__ inline void tdm_load_2d_bf16(unsigned lds_off, const void* gptr,
                                        unsigned tile_d0, unsigned tile_d1,
                                        unsigned long long stride0_elems,
                                        unsigned long long dim0,
                                        unsigned long long dim1) {
  unsigned long long ga = (unsigned long long)gptr;
  v4u g0;
  g0[0] = 1u;                                                 // count=1, user desc
  g0[1] = lds_off;                                            // lds_addr
  g0[2] = (unsigned)(ga & 0xFFFFFFFFu);                       // global_addr[31:0]
  g0[3] = (unsigned)((ga >> 32) & 0x01FFFFFFu) | 0x80000000u; // addr[56:32] | type=2
  v8i g1;
  g1[0] = (int)(1u << 16);                                    // data_size=1 (2B)
  g1[1] = (int)((dim0 & 0xFFFFull) << 16);                    // tensor_dim0 lo
  g1[2] = (int)(((dim0 >> 16) & 0xFFFFull) | ((dim1 & 0xFFFFull) << 16));
  g1[3] = (int)(((dim1 >> 16) & 0xFFFFull) | ((unsigned long long)(tile_d0 & 0xFFFFu) << 16));
  g1[4] = (int)(tile_d1 & 0xFFFFu);                           // tile_dim1 (tile_dim2=0)
  g1[5] = (int)(stride0_elems & 0xFFFFFFFFull);               // tensor_dim0_stride lo
  g1[6] = (int)((stride0_elems >> 32) & 0xFFFFull);           // stride hi
  g1[7] = 0;
  v4i z4 = {0, 0, 0, 0};
#if __clang_major__ >= 23
  v8i z8 = {0, 0, 0, 0, 0, 0, 0, 0};
  __builtin_amdgcn_tensor_load_to_lds(g0, g1, z4, z4, z8, 0);
#else
  __builtin_amdgcn_tensor_load_to_lds(g0, g1, z4, z4, 0);
#endif
}
#endif

// WMMA A-fragment (16x32 bf16) from row-major LDS tile, row stride KT elems.
// Lane l: row l%16, K = kk + (l/16)*16 + 0..15.
__device__ inline v16bf frag_a_lds(const __bf16* tileBase, int rowInTile,
                                   int khalf, int kk) {
  const __bf16* p = tileBase + rowInTile * KT + kk + khalf * 16;
  v8bf lo = *(const v8bf*)(p);
  v8bf hi = *(const v8bf*)(p + 8);
  return __builtin_shufflevector(lo, hi, 0, 1, 2, 3, 4, 5, 6, 7,
                                 8, 9, 10, 11, 12, 13, 14, 15);
}

// WMMA B-fragment from row-major LDS tile (rows = N index, stride KT elems).
// Lane l: col l%16, K = kk + (l/16)*16 + 0..15 (one aligned 32B read).
__device__ inline v16bf frag_b_lds(const __bf16* tileBase, int rowInTile,
                                   int khalf, int kk) {
  return *(const v16bf*)(tileBase + rowInTile * KT + kk + khalf * 16);
}

// ---------------------------------------------------------------------------
// Kernel 0: init transposed labels LT[64][8192] + zero degree array
// ---------------------------------------------------------------------------
__global__ void lp_init_kernel(const int* __restrict__ labels,
                               __bf16* __restrict__ LT0,
                               float* __restrict__ deg) {
  int idx = blockIdx.x * 256 + threadIdx.x;
  if (idx < N_ALL) deg[idx] = 0.0f;
  if (idx < N_ALL * N_CLASSES) {
    int n = idx >> 13;        // class 0..63
    int m = idx & (N_ALL - 1);
    float v = 0.0f;
    if (m < N_SUPPORT) v = (labels[m] == n) ? 1.0f : 0.0f;
    LT0[idx] = (__bf16)v;
  }
}

// ---------------------------------------------------------------------------
// Kernel 1: E = [support;query] @ W   (f32 in, bf16 out), WMMA bf16
// ---------------------------------------------------------------------------
__global__ void lp_embed_kernel(const float* __restrict__ support,
                                const float* __restrict__ query,
                                const float* __restrict__ W,
                                __bf16* __restrict__ E) {
  const int lane  = threadIdx.x & 31;
  const int wave  = threadIdx.x >> 5;
  const int task  = blockIdx.x * 8 + wave;       // 8192 tiles
  const int tm    = task >> 4;                   // 0..511
  const int tn    = task & 15;                   // 0..15
  const int l15   = lane & 15;
  const int khalf = lane >> 4;

  const int row = tm * 16 + l15;
  const float* X = (row < N_SUPPORT) ? (support + (size_t)row * D_IN)
                                     : (query + (size_t)(row - N_SUPPORT) * D_IN);
  v8f acc = {};
  for (int k0 = 0; k0 < D_IN; k0 += 32) {
    v16bf a;
    {
      const float* p = X + k0 + khalf * 16;
#pragma unroll
      for (int i = 0; i < 8; ++i) {
        a[i]     = (__bf16)p[i];
        a[8 + i] = (__bf16)p[8 + i];
      }
    }
    v16bf b;
    {
      const int col = tn * 16 + l15;
      const int kb  = k0 + khalf * 16;
#pragma unroll
      for (int v = 0; v < 8; ++v) {
        b[2 * v]     = (__bf16)W[(size_t)(kb + 2 * v) * D_EMB + col];
        b[2 * v + 1] = (__bf16)W[(size_t)(kb + 2 * v + 1) * D_EMB + col];
      }
    }
    acc = __builtin_amdgcn_wmma_f32_16x16x32_bf16(false, a, false, b,
                                                  (short)0, acc, false, false);
  }
#pragma unroll
  for (int r = 0; r < 8; ++r) {
    int m = tm * 16 + khalf * 8 + r;
    int n = tn * 16 + l15;
    E[(size_t)m * D_EMB + n] = (__bf16)acc[r];
  }
}

// ---------------------------------------------------------------------------
// Kernel 2: S = E @ E^T, adjacency = (S > 0.5) bf16, row degrees.
// Block (8 waves) -> 128x64 tile. Double-buffered TDM pipeline over K.
// ---------------------------------------------------------------------------
__global__ void lp_sim_kernel(const __bf16* __restrict__ E,
                              __bf16* __restrict__ Adj,
                              float* __restrict__ deg) {
  __shared__ __attribute__((aligned(128))) __bf16 sA[2][128 * KT];  // 32 KB
  __shared__ __attribute__((aligned(128))) __bf16 sB[2][64 * KT];   // 16 KB

  const int lane  = threadIdx.x & 31;
  const int wave  = threadIdx.x >> 5;
  const int l15   = lane & 15;
  const int khalf = lane >> 4;
  const int mBase = blockIdx.y * 128;
  const int nBase = blockIdx.x * 64;
  const int T     = D_EMB / KT;   // 4 K-tiles

  v8f acc[4] = {};
#ifdef HAVE_TDM
  if (threadIdx.x < 32) {   // prologue: stage tile 0
    tdm_load_2d_bf16((unsigned)(unsigned long long)(uintptr_t)&sA[0][0],
                     E + (size_t)mBase * D_EMB, KT, 128, D_EMB, D_EMB, N_ALL);
    tdm_load_2d_bf16((unsigned)(unsigned long long)(uintptr_t)&sB[0][0],
                     E + (size_t)nBase * D_EMB, KT, 64, D_EMB, D_EMB, N_ALL);
  }
#endif
  for (int t = 0; t < T; ++t) {
    const int k0 = t * KT;
#ifdef HAVE_TDM
    if (threadIdx.x < 32) {
      if (t + 1 < T) {   // issue next tile, then wait for current (in-order TDM)
        tdm_load_2d_bf16((unsigned)(unsigned long long)(uintptr_t)&sA[(t + 1) & 1][0],
                         E + (size_t)mBase * D_EMB + k0 + KT, KT, 128, D_EMB, D_EMB, N_ALL);
        tdm_load_2d_bf16((unsigned)(unsigned long long)(uintptr_t)&sB[(t + 1) & 1][0],
                         E + (size_t)nBase * D_EMB + k0 + KT, KT, 64, D_EMB, D_EMB, N_ALL);
        __builtin_amdgcn_s_wait_tensorcnt(2);
      } else {
        __builtin_amdgcn_s_wait_tensorcnt(0);
      }
    }
#else
    for (int idx = threadIdx.x; idx < (128 * KT) / 2; idx += 256) {
      int e = idx * 2, r = e / KT, c = e % KT;
      *(unsigned*)&sA[t & 1][e] = *(const unsigned*)&E[(size_t)(mBase + r) * D_EMB + k0 + c];
    }
    for (int idx = threadIdx.x; idx < (64 * KT) / 2; idx += 256) {
      int e = idx * 2, r = e / KT, c = e % KT;
      *(unsigned*)&sB[t & 1][e] = *(const unsigned*)&E[(size_t)(nBase + r) * D_EMB + k0 + c];
    }
#endif
    __syncthreads();
    const __bf16* tA = &sA[t & 1][0];
    const __bf16* tB = &sB[t & 1][0];
#pragma unroll
    for (int kk = 0; kk < KT; kk += 32) {
      v16bf a = frag_a_lds(tA, wave * 16 + l15, khalf, kk);
#pragma unroll
      for (int j = 0; j < 4; ++j) {
        v16bf b = frag_b_lds(tB, j * 16 + l15, khalf, kk);
        acc[j] = __builtin_amdgcn_wmma_f32_16x16x32_bf16(false, a, false, b,
                                                         (short)0, acc[j], false, false);
      }
    }
    __syncthreads();   // buffer (t&1) free for tile t+2
  }

  float rowsum[8];
#pragma unroll
  for (int r = 0; r < 8; ++r) rowsum[r] = 0.0f;

#pragma unroll
  for (int j = 0; j < 4; ++j) {
#pragma unroll
    for (int r = 0; r < 8; ++r) {
      float v = (acc[j][r] > 0.5f) ? 1.0f : 0.0f;
      int m = mBase + wave * 16 + khalf * 8 + r;
      int n = nBase + j * 16 + l15;
      Adj[(size_t)m * N_ALL + n] = (__bf16)v;
      rowsum[r] += v;
    }
  }
#pragma unroll
  for (int r = 0; r < 8; ++r) {
    float s = rowsum[r];
    s += __shfl_xor(s, 1, 32);
    s += __shfl_xor(s, 2, 32);
    s += __shfl_xor(s, 4, 32);
    s += __shfl_xor(s, 8, 32);
    if (l15 == 0) atomicAdd(&deg[mBase + wave * 16 + khalf * 8 + r], s);
  }
}

// ---------------------------------------------------------------------------
// Kernel 3: degInv = 1 / (deg == 0 ? 1 : deg)
// ---------------------------------------------------------------------------
__global__ void lp_recip_kernel(const float* __restrict__ deg,
                                float* __restrict__ degInv) {
  int i = blockIdx.x * 256 + threadIdx.x;
  if (i < N_ALL) {
    float d = deg[i];
    degInv[i] = 1.0f / ((d == 0.0f) ? 1.0f : d);
  }
}

// ---------------------------------------------------------------------------
// Kernel 4: LTout = 0.5*LTin + 0.5*degInv ⊙ (Adj @ L)   (labels transposed)
// Block (8 waves) -> 128 rows x 64 classes. Double-buffered TDM pipeline:
// while WMMAs consume tile t, the Tensor Data Mover fills tile t+1.
// ---------------------------------------------------------------------------
__global__ void lp_prop_kernel(const __bf16* __restrict__ Adj,
                               const __bf16* __restrict__ LTin,
                               const float* __restrict__ degInv,
                               __bf16* __restrict__ LTout) {
  __shared__ __attribute__((aligned(128))) __bf16 sA[2][128 * KT];   // 32 KB
  __shared__ __attribute__((aligned(128))) __bf16 sLT[2][64 * KT];   // 16 KB

  const int lane  = threadIdx.x & 31;
  const int wave  = threadIdx.x >> 5;
  const int l15   = lane & 15;
  const int khalf = lane >> 4;
  const int mBase = blockIdx.x * 128;
  const int T     = N_ALL / KT;   // 128 K-tiles

  v8f acc[4] = {};
#ifdef HAVE_TDM
  if (threadIdx.x < 32) {   // prologue: stage tile 0
    tdm_load_2d_bf16((unsigned)(unsigned long long)(uintptr_t)&sA[0][0],
                     Adj + (size_t)mBase * N_ALL, KT, 128, N_ALL, N_ALL, N_ALL);
    tdm_load_2d_bf16((unsigned)(unsigned long long)(uintptr_t)&sLT[0][0],
                     LTin, KT, 64, N_ALL, N_ALL, N_CLASSES);
  }
#endif
  for (int t = 0; t < T; ++t) {
    const int k0 = t * KT;
#ifdef HAVE_TDM
    if (threadIdx.x < 32) {
      if (t + 1 < T) {
        tdm_load_2d_bf16((unsigned)(unsigned long long)(uintptr_t)&sA[(t + 1) & 1][0],
                         Adj + (size_t)mBase * N_ALL + k0 + KT, KT, 128, N_ALL, N_ALL, N_ALL);
        tdm_load_2d_bf16((unsigned)(unsigned long long)(uintptr_t)&sLT[(t + 1) & 1][0],
                         LTin + k0 + KT, KT, 64, N_ALL, N_ALL, N_CLASSES);
        __builtin_amdgcn_s_wait_tensorcnt(2);   // tile t landed (TDM is in-order)
      } else {
        __builtin_amdgcn_s_wait_tensorcnt(0);
      }
    }
#else
    for (int idx = threadIdx.x; idx < (128 * KT) / 2; idx += 256) {
      int e = idx * 2, r = e / KT, c = e % KT;
      *(unsigned*)&sA[t & 1][e] = *(const unsigned*)&Adj[(size_t)(mBase + r) * N_ALL + k0 + c];
    }
    for (int idx = threadIdx.x; idx < (64 * KT) / 2; idx += 256) {
      int e = idx * 2, r = e / KT, c = e % KT;
      *(unsigned*)&sLT[t & 1][e] = *(const unsigned*)&LTin[(size_t)r * N_ALL + k0 + c];
    }
#endif
    __syncthreads();
    const __bf16* tA = &sA[t & 1][0];
    const __bf16* tL = &sLT[t & 1][0];
#pragma unroll
    for (int kk = 0; kk < KT; kk += 32) {
      v16bf a = frag_a_lds(tA, wave * 16 + l15, khalf, kk);
#pragma unroll
      for (int j = 0; j < 4; ++j) {
        v16bf b = frag_b_lds(tL, j * 16 + l15, khalf, kk);
        acc[j] = __builtin_amdgcn_wmma_f32_16x16x32_bf16(false, a, false, b,
                                                         (short)0, acc[j], false, false);
      }
    }
    __syncthreads();   // buffer (t&1) free for tile t+2
  }

  // Epilogue: rows m0..m0+7 contiguous in LT for fixed class n -> 128-bit ops.
  const int m0 = mBase + wave * 16 + khalf * 8;
  v8f dgi = *(const v8f*)(degInv + m0);
#pragma unroll
  for (int j = 0; j < 4; ++j) {
    int n = j * 16 + l15;
    v8bf oldv = *(const v8bf*)(LTin + (size_t)n * N_ALL + m0);
    v8bf outv;
#pragma unroll
    for (int r = 0; r < 8; ++r) {
      float res = 0.5f * (float)oldv[r] + 0.5f * dgi[r] * acc[j][r];
      outv[r] = (__bf16)res;
    }
    *(v8bf*)(LTout + (size_t)n * N_ALL + m0) = outv;
  }
}

// ---------------------------------------------------------------------------
// Kernel 5: query labels (transposed) -> f32 row-major output
// ---------------------------------------------------------------------------
__global__ void lp_out_kernel(const __bf16* __restrict__ LT,
                              float* __restrict__ out) {
  int i = blockIdx.x * 256 + threadIdx.x;
  if (i < N_QUERY * N_CLASSES) {
    int q = i >> 6;          // query row
    int n = i & 63;          // class
    out[i] = (float)LT[(size_t)n * N_ALL + N_SUPPORT + q];
  }
}

// ---------------------------------------------------------------------------
extern "C" void kernel_launch(void* const* d_in, const int* in_sizes, int n_in,
                              void* d_out, int out_size, void* d_ws, size_t ws_size,
                              hipStream_t stream) {
  const float* support = (const float*)d_in[0];
  const float* query   = (const float*)d_in[1];
  const float* W       = (const float*)d_in[2];
  const int*   labels  = (const int*)d_in[3];
  float*       out     = (float*)d_out;

  char* ws = (char*)d_ws;
  size_t off = 0;
  __bf16* Adj = (__bf16*)(ws + off); off += (size_t)N_ALL * N_ALL * 2;      // 134 MB
  __bf16* E   = (__bf16*)(ws + off); off += (size_t)N_ALL * D_EMB * 2;      // 4 MB
  __bf16* LT0 = (__bf16*)(ws + off); off += (size_t)N_ALL * N_CLASSES * 2;  // 1 MB
  __bf16* LT1 = (__bf16*)(ws + off); off += (size_t)N_ALL * N_CLASSES * 2;  // 1 MB
  float*  deg = (float*)(ws + off);  off += (size_t)N_ALL * 4;
  float*  dgi = (float*)(ws + off);  off += (size_t)N_ALL * 4;

  lp_init_kernel<<<(N_ALL * N_CLASSES) / 256, 256, 0, stream>>>(labels, LT0, deg);
  lp_embed_kernel<<<1024, 256, 0, stream>>>(support, query, W, E);
  lp_sim_kernel<<<dim3(N_ALL / 64, N_ALL / 128), 256, 0, stream>>>(E, Adj, deg);
  lp_recip_kernel<<<N_ALL / 256, 256, 0, stream>>>(deg, dgi);
  __bf16* bufs[2] = {LT0, LT1};
  for (int it = 0; it < ITERS; ++it) {
    lp_prop_kernel<<<N_ALL / 128, 256, 0, stream>>>(Adj, bufs[it & 1], dgi,
                                                    bufs[(it + 1) & 1]);
  }
  lp_out_kernel<<<(N_QUERY * N_CLASSES) / 256, 256, 0, stream>>>(bufs[ITERS & 1], out);
}
